// SIGRegLoss_28664611733879
// MI455X (gfx1250) — matrix operand order repaired and checked by reading, not compiled, not run on hardware
//
#include <hip/hip_runtime.h>
#include <stdint.h>

typedef __attribute__((ext_vector_type(16))) _Float16     v16h;
typedef __attribute__((ext_vector_type(8)))  float        v8f;
typedef __attribute__((ext_vector_type(4)))  unsigned int v4u;
typedef __attribute__((ext_vector_type(8)))  int          v8i;
typedef __attribute__((ext_vector_type(4)))  int          v4i;

#define N_SAMP 1024   // B*L
#define D_DIM  1024
#define S_SK   3
#define K_SK   128

#define TILE_N 64     // rows per block (4 waves x 16)
#define TILE_D 32     // contraction chunk (one WMMA K)

#if defined(__has_builtin)
#if __has_builtin(__builtin_amdgcn_tensor_load_to_lds)
#define SIGREG_HAVE_TDM 1
#endif
#endif
#if __has_include(<hip/amd_detail/amd_gfx1250_TDM.h>)
#define SIGREG_TDM_6ARG 1
#endif

// ---------------------------------------------------------------------------
// Kernel 0: zero the scalar output (graph-capture safe init)
// ---------------------------------------------------------------------------
__global__ void sigreg_zero(float* __restrict__ out) {
    if (threadIdx.x == 0 && blockIdx.x == 0) out[0] = 0.0f;
}

#if defined(SIGREG_HAVE_TDM)
// ---------------------------------------------------------------------------
// Issue a TDM 2-D tile load: global (row_stride pitch, elements) -> LDS,
// tile_w x tile_h f32 elements, packed row-major contiguously in LDS.
// D# layout per CDNA5 ISA ch.8: group0 = {count/type, lds_addr, global_addr},
// group1 = {data_size, tensor dims, tile dims, dim0 stride}. Groups 2/3 zero
// (2-D tensor). Tracked by TENSORcnt.
// ---------------------------------------------------------------------------
__device__ __forceinline__ void tdm_load_tile_2d(const float* gsrc,
                                                 uint32_t lds_byte,
                                                 uint32_t tile_w,
                                                 uint32_t tile_h,
                                                 uint32_t row_stride_elems)
{
    const uint64_t ga = (uint64_t)(uintptr_t)gsrc;
    v4u g0;
    g0[0] = 1u;                                   // count=1, user descriptor
    g0[1] = lds_byte;                             // lds_addr (bytes)
    g0[2] = (uint32_t)ga;                         // global_addr[31:0]  (bits 95:64)
    g0[3] = (uint32_t)((ga >> 32) & 0x01ffffffu)  // global_addr[56:32] (bits 120:96)
          | (2u << 30);                           // type=2 ("image")
    v8i g1;
    g1[0] = (int)(2u << 16);                      // wg_mask=0, data_size=2 (4 bytes)
    g1[1] = (int)((row_stride_elems & 0xffffu) << 16);            // tensor_dim0 lo
    g1[2] = (int)((row_stride_elems >> 16) | ((tile_h & 0xffffu) << 16)); // dim0 hi | tensor_dim1 lo
    g1[3] = (int)((tile_h >> 16) | (tile_w << 16));               // dim1 hi | tile_dim0
    g1[4] = (int)(tile_h & 0xffffu);              // tile_dim1 | tile_dim2=0
    g1[5] = (int)row_stride_elems;                // tensor_dim0_stride[31:0]
    g1[6] = 0;                                    // stride hi | dim1_stride lo
    g1[7] = 0;                                    // dim1_stride hi
    v4i z4 = {0, 0, 0, 0};
#if defined(SIGREG_TDM_6ARG)
    v8i z8 = {0, 0, 0, 0, 0, 0, 0, 0};
    __builtin_amdgcn_tensor_load_to_lds(g0, g1, z4, z4, z8, 0);
#else
    __builtin_amdgcn_tensor_load_to_lds(g0, g1, z4, z4, 0);
#endif
}
#endif // SIGREG_HAVE_TDM

// ---------------------------------------------------------------------------
// Kernel 1: ep[s][k][n] = sum_d emb[n][d]   * proj[s][d][k] / sqrt(D)
//           rp[s][k][n] = sum_d noise[n][d] * proj[s][d][k] / sqrt(D)
// 128-thread block (4 waves) computes a 64x16 (n,k) tile for both outputs.
// TDM double-buffers A/B tiles into LDS; waves read fragments from LDS and
// issue v_wmma_f32_16x16x32_f16 (f16 in, f32 accumulate).
// ---------------------------------------------------------------------------
__global__ __launch_bounds__(128)
void sigreg_project_wmma(const float* __restrict__ emb,
                         const float* __restrict__ noise,
                         const float* __restrict__ proj,
                         float* __restrict__ ep,
                         float* __restrict__ rp)
{
    __shared__ float ldsE[2][TILE_N * TILE_D];   // 2 x 8 KB
    __shared__ float ldsR[2][TILE_N * TILE_D];   // 2 x 8 KB
    __shared__ float ldsP[2][TILE_D * 16];       // 2 x 2 KB

    const int tid  = threadIdx.x;
    const int lane = tid & 31;
    const int wave = tid >> 5;
    const int row  = lane & 15;        // A row / B,C column within tile
    const int half = lane >> 4;        // 0 or 1

    const int k0 = blockIdx.x * 16;
    const int n0 = blockIdx.y * TILE_N;
    const int s  = blockIdx.z;

    const float* embT = emb   + (size_t)n0 * D_DIM;
    const float* noiT = noise + (size_t)n0 * D_DIM;
    const float* prjT = proj  + (size_t)s * D_DIM * K_SK + k0;

    auto stage = [&](int buf, int d0) {
#if defined(SIGREG_HAVE_TDM)
        if (wave == 0) {               // TDM ignores EXEC; one wave issues DMA
            tdm_load_tile_2d(embT + d0,
                             (uint32_t)(uintptr_t)(void*)&ldsE[buf][0],
                             TILE_D, TILE_N, D_DIM);
            tdm_load_tile_2d(noiT + d0,
                             (uint32_t)(uintptr_t)(void*)&ldsR[buf][0],
                             TILE_D, TILE_N, D_DIM);
            tdm_load_tile_2d(prjT + (size_t)d0 * K_SK,
                             (uint32_t)(uintptr_t)(void*)&ldsP[buf][0],
                             16, TILE_D, K_SK);
        }
#else
        // Fallback: cooperative float4 staging
        for (int i = tid; i < (TILE_N * TILE_D) / 4; i += 128) {
            const int r = i / (TILE_D / 4);
            const int c = i % (TILE_D / 4);
            ((float4*)&ldsE[buf][r * TILE_D])[c] =
                ((const float4*)(embT + (size_t)r * D_DIM + d0))[c];
            ((float4*)&ldsR[buf][r * TILE_D])[c] =
                ((const float4*)(noiT + (size_t)r * D_DIM + d0))[c];
        }
        for (int i = tid; i < (TILE_D * 16) / 4; i += 128) {
            const int r = i / 4;
            const int c = i % 4;
            ((float4*)&ldsP[buf][r * 16])[c] =
                ((const float4*)(prjT + (size_t)(d0 + r) * K_SK))[c];
        }
#endif
    };

    const float scale = 0.03125f;      // 1/sqrt(1024), folded into B
    v8f cE = {};
    v8f cR = {};

    stage(0, 0);                       // prologue: chunk 0 into buffer 0

    constexpr int NCHUNK = D_DIM / TILE_D;   // 32
    for (int it = 0; it < NCHUNK; ++it) {
        const int buf = it & 1;
        if (it + 1 < NCHUNK) {
            stage(buf ^ 1, (it + 1) * TILE_D);   // overlap next DMA
#if defined(SIGREG_HAVE_TDM)
            __builtin_amdgcn_s_wait_tensorcnt(3); // chunk 'it' (3 ops) landed
#endif
        } else {
#if defined(SIGREG_HAVE_TDM)
            __builtin_amdgcn_s_wait_tensorcnt(0);
#endif
        }
        __syncthreads();

        // A fragments (16x32 f16) from LDS: lanes 0-15 row holds K={0..7,16..23},
        // lanes 16-31 hold K={8..15,24..31} (ISA 7.12.2)
        const float* Ae = &ldsE[buf][(wave * 16 + row) * TILE_D];
        const float* Ar = &ldsR[buf][(wave * 16 + row) * TILE_D];
        const int klo = half * 8;
        v16h aE, aR, b;
#pragma unroll
        for (int t = 0; t < 8; ++t) {
            aE[t]     = (_Float16)Ae[klo + t];
            aE[8 + t] = (_Float16)Ae[16 + klo + t];
            aR[t]     = (_Float16)Ar[klo + t];
            aR[8 + t] = (_Float16)Ar[16 + klo + t];
        }
        // B fragment (32x16 f16): lane holds column k0+row; lanes 0-15 rows
        // d0+{0..7,16..23}, lanes 16-31 rows d0+{8..15,24..31}
#pragma unroll
        for (int t = 0; t < 16; ++t)
            b[t] = (_Float16)(ldsP[buf][((t & 7) + half * 8 + (t >> 3) * 16) * 16 + row] * scale);

        cE = __builtin_amdgcn_wmma_f32_16x16x32_f16(
                 false, aE, false, b, (short)0, cE, false, false);
        cR = __builtin_amdgcn_wmma_f32_16x16x32_f16(
                 false, aR, false, b, (short)0, cR, false, false);

        __syncthreads();               // protect buf^1 before next DMA issue
    }

    // C/D layout: VGPR v -> (M=v, N=lane) lanes 0-15; (M=v+8, N=lane-16) lanes 16-31
    const int mofs  = half * 8;
    const int nbase = n0 + wave * 16;
    float* eout = ep + ((size_t)s * K_SK + (size_t)(k0 + row)) * N_SAMP + nbase;
    float* rout = rp + ((size_t)s * K_SK + (size_t)(k0 + row)) * N_SAMP + nbase;
#pragma unroll
    for (int v = 0; v < 8; ++v) {
        eout[mofs + v] = cE[v];
        rout[mofs + v] = cR[v];
    }
}

// ---------------------------------------------------------------------------
// Kernel 2: per-(s,k) energy-distance statistic, squared, accumulated.
// stat = (2*Sxy - Sxx - Syy)/N^2 ; out += stat^2 / (S*K)
// One block of 256 threads per (s,k); x,y resident in LDS (8 KB).
// ---------------------------------------------------------------------------
__global__ __launch_bounds__(256)
void sigreg_energy_stat(const float* __restrict__ ep,
                        const float* __restrict__ rp,
                        float* __restrict__ out)
{
    __shared__ float xs[N_SAMP];
    __shared__ float ys[N_SAMP];
    __shared__ float red[256];

    const int tid = threadIdx.x;
    const int blk = blockIdx.x;                 // s*K + k
    const float* xsrc = ep + (size_t)blk * N_SAMP;
    const float* ysrc = rp + (size_t)blk * N_SAMP;

    for (int i = tid; i < N_SAMP; i += 256) {
        xs[i] = xsrc[i];
        ys[i] = ysrc[i];
    }
    __syncthreads();

    const float x0 = xs[tid],       y0 = ys[tid];
    const float x1 = xs[tid + 256], y1 = ys[tid + 256];
    const float x2 = xs[tid + 512], y2 = ys[tid + 512];
    const float x3 = xs[tid + 768], y3 = ys[tid + 768];

    float sxx = 0.0f, syy = 0.0f, sxy = 0.0f;
#pragma unroll 4
    for (int j = 0; j < N_SAMP; ++j) {
        const float xj = xs[j];
        const float yj = ys[j];
        sxx += fabsf(x0 - xj) + fabsf(x1 - xj) + fabsf(x2 - xj) + fabsf(x3 - xj);
        syy += fabsf(y0 - yj) + fabsf(y1 - yj) + fabsf(y2 - yj) + fabsf(y3 - yj);
        sxy += fabsf(x0 - yj) + fabsf(x1 - yj) + fabsf(x2 - yj) + fabsf(x3 - yj);
    }

    // stat is linear in the partial sums -> reduce the combination once
    red[tid] = 2.0f * sxy - sxx - syy;
    __syncthreads();
#pragma unroll
    for (int off = 128; off > 0; off >>= 1) {
        if (tid < off) red[tid] += red[tid + off];
        __syncthreads();
    }

    if (tid == 0) {
        const float inv_n2 = 1.0f / ((float)N_SAMP * (float)N_SAMP);
        const float stat   = red[0] * inv_n2;
        atomicAdd(out, stat * stat * (1.0f / (float)(S_SK * K_SK)));
    }
}

// ---------------------------------------------------------------------------
extern "C" void kernel_launch(void* const* d_in, const int* in_sizes, int n_in,
                              void* d_out, int out_size, void* d_ws, size_t ws_size,
                              hipStream_t stream)
{
    const float* emb   = (const float*)d_in[0];   // [8,128,1024] f32
    const float* noise = (const float*)d_in[1];   // [1024,1024]  f32
    const float* proj  = (const float*)d_in[2];   // [3,1024,128] f32
    float* out = (float*)d_out;                   // scalar f32

    float* ep = (float*)d_ws;                               // [S,K,N]
    float* rp = ep + (size_t)S_SK * K_SK * N_SAMP;          // [S,K,N]

    sigreg_zero<<<1, 32, 0, stream>>>(out);

    dim3 g1(K_SK / 16, N_SAMP / TILE_N, S_SK);              // 8 x 16 x 3 blocks
    sigreg_project_wmma<<<g1, 128, 0, stream>>>(emb, noise, proj, ep, rp);

    sigreg_energy_stat<<<S_SK * K_SK, 256, 0, stream>>>(ep, rp, out);
}
